// MyLlamaAttention_3341484557030
// MI455X (gfx1250) — matrix operand order repaired and checked by reading, not compiled, not run on hardware
//
#include <hip/hip_runtime.h>
#include <hip/hip_bf16.h>

// ---------------- types ----------------
typedef __bf16 bf16;
typedef __attribute__((ext_vector_type(16))) __bf16 v16bf;
typedef __attribute__((ext_vector_type(8)))  __bf16 v8bf;
typedef __attribute__((ext_vector_type(8)))  float  v8f;
typedef __attribute__((ext_vector_type(4)))  float  v4f;

// ---------------- problem constants ----------------
#define S_LEN 2048
#define HID   2048
#define NH    32
#define NKV   8
#define HD    64
#define NQKV  3072   // NH*HD + 2*NKV*HD
#define NEGV  (-1.0e9f)
#define BROW  40     // padded LDS row stride (elements) for 32-wide B slab

static __device__ inline v16bf cat8(v8bf lo, v8bf hi) {
    return __builtin_shufflevector(lo, hi, 0,1,2,3,4,5,6,7,8,9,10,11,12,13,14,15);
}
static __device__ inline v8f vzero8() {
    v8f z;
#pragma unroll
    for (int i = 0; i < 8; ++i) z[i] = 0.0f;
    return z;
}
static __device__ inline v8f wmma_bf16(v16bf a, v16bf b, v8f c) {
    return __builtin_amdgcn_wmma_f32_16x16x32_bf16(false, a, false, b, (short)0, c, false, false);
}
static __device__ inline v8bf cvt8(const float* __restrict__ p) {
    const v4f* q = (const v4f*)p;
    v4f x0 = q[0], x1 = q[1];
    v8bf r;
#pragma unroll
    for (int j = 0; j < 4; ++j) { r[j] = (bf16)x0[j]; r[4 + j] = (bf16)x1[j]; }
    return r;
}

// ---- CDNA5 async global->LDS copy (no VGPR round-trip, tracked by ASYNCcnt) ----
static __device__ inline void async_copy_b128(const bf16* gsrc, bf16* ldst) {
    const unsigned lds_addr = (unsigned)(size_t)ldst;   // LDS aperture: addr[31:0]
    asm volatile("global_load_async_to_lds_b128 %0, %1, off"
                 :: "v"(lds_addr), "v"(gsrc) : "memory");
}
static __device__ inline void wait_asynccnt0() {
    asm volatile("s_wait_asynccnt 0x0" ::: "memory");
}

// ---------------- kernel 1: fp32 -> bf16 conversions (vectorized) ----------------
__global__ __launch_bounds__(256) void convert_kernel(
    const float* __restrict__ h,  const float* __restrict__ wq,
    const float* __restrict__ wk, const float* __restrict__ wv,
    const float* __restrict__ wo,
    bf16* __restrict__ hb, bf16* __restrict__ wqkvb, bf16* __restrict__ wob)
{
    const size_t stride = (size_t)gridDim.x * blockDim.x;
    const size_t t0 = (size_t)blockIdx.x * blockDim.x + threadIdx.x;
    const size_t NBIG  = (size_t)S_LEN * HID;      // 4,194,304
    const size_t NBIG8 = NBIG / 8;
    for (size_t i = t0; i < NBIG8; i += stride) {
        const size_t o = i * 8;
        ((v8bf*)hb)[i]    = cvt8(h + o);
        ((v8bf*)wob)[i]   = cvt8(wo + o);
        ((v8bf*)wqkvb)[i] = cvt8(wq + o);
    }
    const size_t KVSZ  = (size_t)NKV * HD * HID;   // 1,048,576
    const size_t KVSZ8 = KVSZ / 8;
    v8bf* wk_dst = (v8bf*)(wqkvb + NBIG);
    v8bf* wv_dst = (v8bf*)(wqkvb + NBIG + KVSZ);
    for (size_t i = t0; i < KVSZ8; i += stride) {
        wk_dst[i] = cvt8(wk + i * 8);
        wv_dst[i] = cvt8(wv + i * 8);
    }
}

// ---------------- kernel 2: QKV projection + fused RoPE ----------------
// Block = 8 waves, tile 256(M) x 64(N). Wave w owns rows {w*16, 128+w*16}.
// B slab (64 x 32 k) double-buffered in LDS via async global->LDS DMA.
__global__ __launch_bounds__(256) void qkv_rope_kernel(
    const bf16* __restrict__ Hb, const bf16* __restrict__ Wqkvb,
    const float* __restrict__ cosT, const float* __restrict__ sinT,
    bf16* __restrict__ Qb, bf16* __restrict__ Kb, bf16* __restrict__ Vtb)
{
    __shared__ bf16 bT[2][64 * BROW];                 // 2 x 5 KB

    const int tid  = threadIdx.x;
    const int lane = tid & 31;
    const int lm   = lane & 15;
    const int half = lane >> 4;
    const int w    = tid >> 5;

    const int nBlk = NQKV / 64;                       // 48
    const int bm = blockIdx.x / nBlk;                 // 0..7  (M/256)
    const int bn = blockIdx.x % nBlk;
    const int n0 = bn * 64;

    const int brow = tid >> 2, bchunk = tid & 3;
    const int bOff = brow * BROW + bchunk * 8;
    const bf16* bSrc = Wqkvb + (size_t)(n0 + brow) * HID + bchunk * 8;

    const bf16* aRow[2];
    int m0[2];
#pragma unroll
    for (int u = 0; u < 2; ++u) {
        m0[u] = bm * 256 + u * 128 + w * 16;
        aRow[u] = Hb + (size_t)(m0[u] + lm) * HID;
    }

    // stage-0 preload: async DMA into bT[0]
    async_copy_b128(bSrc, &bT[0][bOff]);
    v8bf aLoE[2], aHiE[2], aLoO[2], aHiO[2];
#pragma unroll
    for (int u = 0; u < 2; ++u) {
        aLoE[u] = *(const v8bf*)(aRow[u] + 8 * half);
        aHiE[u] = *(const v8bf*)(aRow[u] + 16 + 8 * half);
    }
    wait_asynccnt0();
    __syncthreads();

    v8f acc[2][4];
#pragma unroll
    for (int u = 0; u < 2; ++u)
#pragma unroll
        for (int t = 0; t < 4; ++t) acc[u][t] = vzero8();

    for (int kk = 0; kk < HID; kk += 64) {
        // ======== even stage: consume bT[0] + aE; DMA next slab into bT[1] ========
        if (kk + 264 <= HID)
            __builtin_prefetch((const void*)(bSrc + kk + 256), 0, 3);  // global_prefetch_b8
        async_copy_b128(bSrc + kk + 32, &bT[1][bOff]);
#pragma unroll
        for (int u = 0; u < 2; ++u) {
            aLoO[u] = *(const v8bf*)(aRow[u] + kk + 32 + 8 * half);
            aHiO[u] = *(const v8bf*)(aRow[u] + kk + 48 + 8 * half);
        }
#pragma unroll
        for (int u = 0; u < 2; ++u) {
            const v16bf av = cat8(aLoE[u], aHiE[u]);
#pragma unroll
            for (int t = 0; t < 4; ++t) {
                const v8bf* pb = (const v8bf*)&bT[0][(16 * t + lm) * BROW + 8 * half];
                acc[u][t] = wmma_bf16(av, cat8(pb[0], pb[2]), acc[u][t]);
            }
        }
        wait_asynccnt0();                             // our DMA into bT[1] done
        __syncthreads();                              // everyone's DMA done

        // ======== odd stage: consume bT[1] + aO; DMA next slab into bT[0] ========
        const bool more = (kk + 64) < HID;
        if (more) {
            async_copy_b128(bSrc + kk + 64, &bT[0][bOff]);
#pragma unroll
            for (int u = 0; u < 2; ++u) {
                aLoE[u] = *(const v8bf*)(aRow[u] + kk + 64 + 8 * half);
                aHiE[u] = *(const v8bf*)(aRow[u] + kk + 80 + 8 * half);
            }
        }
#pragma unroll
        for (int u = 0; u < 2; ++u) {
            const v16bf av = cat8(aLoO[u], aHiO[u]);
#pragma unroll
            for (int t = 0; t < 4; ++t) {
                const v8bf* pb = (const v8bf*)&bT[1][(16 * t + lm) * BROW + 8 * half];
                acc[u][t] = wmma_bf16(av, cat8(pb[0], pb[2]), acc[u][t]);
            }
        }
        wait_asynccnt0();
        __syncthreads();
    }

    // ---- fused RoPE on Q and K regions (n0 < 2560). d and d+32 are acc[.][t], acc[.][t+2].
    if (n0 < 2560) {
#pragma unroll
        for (int u = 0; u < 2; ++u)
#pragma unroll
            for (int t = 0; t < 2; ++t) {
                const int d1 = 16 * t + lm;
#pragma unroll
                for (int r = 0; r < 8; ++r) {
                    const int srow = m0[u] + r + 8 * half;
                    const float c1 = cosT[srow * HD + d1];
                    const float s1 = sinT[srow * HD + d1];
                    const float c2 = cosT[srow * HD + d1 + 32];
                    const float s2 = sinT[srow * HD + d1 + 32];
                    const float x1 = acc[u][t][r];
                    const float x2 = acc[u][t + 2][r];
                    acc[u][t][r]     = x1 * c1 - x2 * s1;
                    acc[u][t + 2][r] = x2 * c2 + x1 * s2;
                }
            }
    }

    // ---- scatter
#pragma unroll
    for (int u = 0; u < 2; ++u)
#pragma unroll
        for (int t = 0; t < 4; ++t) {
            const int n = n0 + 16 * t + lm;
#pragma unroll
            for (int r = 0; r < 8; ++r) {
                const int srow = m0[u] + r + 8 * half;
                const bf16 val = (bf16)acc[u][t][r];
                if (n0 < 2048) {                      // Q: [s][nh*64]
                    Qb[(size_t)srow * (NH * HD) + n] = val;
                } else if (n0 < 2560) {               // K: [kv][s][d]
                    const int kvh = (n - 2048) >> 6, d = n & 63;
                    Kb[((size_t)kvh * S_LEN + srow) * HD + d] = val;
                } else {                              // V transposed: [kv][d][s]
                    const int kvh = (n - 2560) >> 6, d = n & 63;
                    Vtb[((size_t)kvh * HD + d) * S_LEN + srow] = val;
                }
            }
        }
}

// ---------------- kernel 3: flash attention (one wave per head x 16-query tile) --------
__global__ __launch_bounds__(32) void attn_kernel(
    const bf16* __restrict__ Qb, const bf16* __restrict__ Kb,
    const bf16* __restrict__ Vtb, bf16* __restrict__ Ab)
{
    __shared__ bf16 pLds[16 * 32];                    // P tile [m][k], 1 KB

    const int lane = threadIdx.x;
    const int lm   = lane & 15;
    const int half = lane >> 4;
    const int h  = blockIdx.x >> 7;
    const int qt = blockIdx.x & 127;
    const int q0 = qt * 16;
    const int kvh = h >> 2;                           // GQA: 4 Q heads per KV head

    const v8bf* pq = (const v8bf*)(Qb + (size_t)(q0 + lm) * (NH * HD) + h * HD);
    const v16bf qa0 = cat8(pq[2 * half],     pq[2 * half + 1]);   // d in [0,32)
    const v16bf qa1 = cat8(pq[4 + 2 * half], pq[5 + 2 * half]);   // d in [32,64)

    v8f oacc[4];
#pragma unroll
    for (int j = 0; j < 4; ++j) oacc[j] = vzero8();
    float mrow[8], lrow[8];
#pragma unroll
    for (int r = 0; r < 8; ++r) { mrow[r] = -1.0e30f; lrow[r] = 0.0f; }

    const bf16* kBase = Kb  + (size_t)kvh * S_LEN * HD;
    const bf16* vBase = Vtb + (size_t)kvh * HD * S_LEN;

    for (int kb = 0; kb < q0 + 16; kb += 32) {
        // deep prefetch of the next K chunk (one line per lane's next key row)
        if (kb + 32 < q0 + 16)
            __builtin_prefetch((const void*)(kBase + (size_t)(kb + 32 + lane) * HD), 0, 3);

        // prefetch V chunk into registers: overlaps score WMMAs + softmax below
        v16bf vb[4];
#pragma unroll
        for (int j = 0; j < 4; ++j) {
            const v8bf* pv = (const v8bf*)(vBase + (size_t)(16 * j + lm) * S_LEN + kb + 16 * half);
            vb[j] = cat8(pv[0], pv[1]);
        }

        // scores S = Q K^T : 16 queries x 32 keys, 4 WMMAs
        v8f sacc[2];
        sacc[0] = vzero8(); sacc[1] = vzero8();
#pragma unroll
        for (int t = 0; t < 2; ++t) {
            const v8bf* pk = (const v8bf*)(kBase + (size_t)(kb + 16 * t + lm) * HD);
            sacc[t] = wmma_bf16(qa0, cat8(pk[2 * half],     pk[2 * half + 1]), sacc[t]);
            sacc[t] = wmma_bf16(qa1, cat8(pk[4 + 2 * half], pk[5 + 2 * half]), sacc[t]);
        }

        // scale + causal mask + online softmax (row = r + 8*half, col = key)
        float scl[8];
#pragma unroll
        for (int r = 0; r < 8; ++r) {
            const int qrow = q0 + r + 8 * half;
#pragma unroll
            for (int t = 0; t < 2; ++t) {
                const int key = kb + 16 * t + lm;
                sacc[t][r] = sacc[t][r] * 0.125f + (key > qrow ? NEGV : 0.0f);
            }
            float cm = fmaxf(sacc[0][r], sacc[1][r]);
            cm = fmaxf(cm, __shfl_xor(cm, 1, 32));
            cm = fmaxf(cm, __shfl_xor(cm, 2, 32));
            cm = fmaxf(cm, __shfl_xor(cm, 4, 32));
            cm = fmaxf(cm, __shfl_xor(cm, 8, 32));
            const float mnew = fmaxf(mrow[r], cm);
            scl[r] = __expf(mrow[r] - mnew);
            const float p0 = __expf(sacc[0][r] - mnew);
            const float p1 = __expf(sacc[1][r] - mnew);
            pLds[(r + 8 * half) * 32 + lm]      = (bf16)p0;   // C-layout -> [m][k]
            pLds[(r + 8 * half) * 32 + 16 + lm] = (bf16)p1;
            float rs = p0 + p1;
            rs += __shfl_xor(rs, 1, 32);
            rs += __shfl_xor(rs, 2, 32);
            rs += __shfl_xor(rs, 4, 32);
            rs += __shfl_xor(rs, 8, 32);
            lrow[r] = lrow[r] * scl[r] + rs;
            mrow[r] = mnew;
        }
#pragma unroll
        for (int j = 0; j < 4; ++j)
#pragma unroll
            for (int r = 0; r < 8; ++r) oacc[j][r] *= scl[r];

        __syncthreads();                              // LDS store -> load order (1-wave WG)

        // P (A layout) from LDS, O += P @ V with prefetched V registers
        const v8bf* pp = (const v8bf*)(pLds + lm * 32);
        const v16bf pa = cat8(pp[2 * half], pp[2 * half + 1]);
#pragma unroll
        for (int j = 0; j < 4; ++j) oacc[j] = wmma_bf16(pa, vb[j], oacc[j]);
        __syncthreads();
    }

#pragma unroll
    for (int j = 0; j < 4; ++j)
#pragma unroll
        for (int r = 0; r < 8; ++r) {
            const float v = oacc[j][r] / lrow[r];
            Ab[(size_t)(q0 + r + 8 * half) * (NH * HD) + h * HD + 16 * j + lm] = (bf16)v;
        }
}

// ---------------- kernel 4: output projection  out = attn @ Wo^T (fp32 out) ----------
__global__ __launch_bounds__(256) void out_gemm_kernel(
    const bf16* __restrict__ Ab, const bf16* __restrict__ Wob, float* __restrict__ out)
{
    __shared__ bf16 bT[2][64 * BROW];

    const int tid  = threadIdx.x;
    const int lane = tid & 31;
    const int lm   = lane & 15;
    const int half = lane >> 4;
    const int w    = tid >> 5;

    const int nBlk = HID / 64;                        // 32
    const int bm = blockIdx.x / nBlk;                 // 0..7
    const int bn = blockIdx.x % nBlk;
    const int n0 = bn * 64;

    const int brow = tid >> 2, bchunk = tid & 3;
    const int bOff = brow * BROW + bchunk * 8;
    const bf16* bSrc = Wob + (size_t)(n0 + brow) * HID + bchunk * 8;

    const bf16* aRow[2];
    int m0[2];
#pragma unroll
    for (int u = 0; u < 2; ++u) {
        m0[u] = bm * 256 + u * 128 + w * 16;
        aRow[u] = Ab + (size_t)(m0[u] + lm) * HID;
    }

    async_copy_b128(bSrc, &bT[0][bOff]);
    v8bf aLoE[2], aHiE[2], aLoO[2], aHiO[2];
#pragma unroll
    for (int u = 0; u < 2; ++u) {
        aLoE[u] = *(const v8bf*)(aRow[u] + 8 * half);
        aHiE[u] = *(const v8bf*)(aRow[u] + 16 + 8 * half);
    }
    wait_asynccnt0();
    __syncthreads();

    v8f acc[2][4];
#pragma unroll
    for (int u = 0; u < 2; ++u)
#pragma unroll
        for (int t = 0; t < 4; ++t) acc[u][t] = vzero8();

    for (int kk = 0; kk < HID; kk += 64) {
        // even stage
        if (kk + 264 <= HID)
            __builtin_prefetch((const void*)(bSrc + kk + 256), 0, 3);
        async_copy_b128(bSrc + kk + 32, &bT[1][bOff]);
#pragma unroll
        for (int u = 0; u < 2; ++u) {
            aLoO[u] = *(const v8bf*)(aRow[u] + kk + 32 + 8 * half);
            aHiO[u] = *(const v8bf*)(aRow[u] + kk + 48 + 8 * half);
        }
#pragma unroll
        for (int u = 0; u < 2; ++u) {
            const v16bf av = cat8(aLoE[u], aHiE[u]);
#pragma unroll
            for (int t = 0; t < 4; ++t) {
                const v8bf* pb = (const v8bf*)&bT[0][(16 * t + lm) * BROW + 8 * half];
                acc[u][t] = wmma_bf16(av, cat8(pb[0], pb[2]), acc[u][t]);
            }
        }
        wait_asynccnt0();
        __syncthreads();

        // odd stage
        const bool more = (kk + 64) < HID;
        if (more) {
            async_copy_b128(bSrc + kk + 64, &bT[0][bOff]);
#pragma unroll
            for (int u = 0; u < 2; ++u) {
                aLoE[u] = *(const v8bf*)(aRow[u] + kk + 64 + 8 * half);
                aHiE[u] = *(const v8bf*)(aRow[u] + kk + 80 + 8 * half);
            }
        }
#pragma unroll
        for (int u = 0; u < 2; ++u) {
            const v16bf av = cat8(aLoO[u], aHiO[u]);
#pragma unroll
            for (int t = 0; t < 4; ++t) {
                const v8bf* pb = (const v8bf*)&bT[1][(16 * t + lm) * BROW + 8 * half];
                acc[u][t] = wmma_bf16(av, cat8(pb[0], pb[2]), acc[u][t]);
            }
        }
        wait_asynccnt0();
        __syncthreads();
    }

#pragma unroll
    for (int u = 0; u < 2; ++u)
#pragma unroll
        for (int t = 0; t < 4; ++t) {
            const int n = n0 + 16 * t + lm;
#pragma unroll
            for (int r = 0; r < 8; ++r) {
                out[(size_t)(m0[u] + r + 8 * half) * HID + n] = acc[u][t][r];
            }
        }
}

// ---------------- host ----------------
extern "C" void kernel_launch(void* const* d_in, const int* in_sizes, int n_in,
                              void* d_out, int out_size, void* d_ws, size_t ws_size,
                              hipStream_t stream) {
    (void)in_sizes; (void)n_in; (void)out_size; (void)ws_size;
    const float* h    = (const float*)d_in[0];
    const float* wq   = (const float*)d_in[1];
    const float* wk   = (const float*)d_in[2];
    const float* wv   = (const float*)d_in[3];
    const float* wo   = (const float*)d_in[4];
    const float* cosT = (const float*)d_in[5];
    const float* sinT = (const float*)d_in[6];
    // d_in[7] (attention_mask) reproduced analytically (causal + NEG).
    float* out = (float*)d_out;

    char* ws = (char*)d_ws;
    const size_t oH    = 0;                                   // Hb : 8 MB (reused as Ab)
    const size_t oWqkv = oH    + (size_t)S_LEN * HID * 2;     // Wqkvb: 12 MB
    const size_t oWo   = oWqkv + (size_t)NQKV * HID * 2;      // Wob: 8 MB
    const size_t oQ    = oWo   + (size_t)HID * HID * 2;       // Qb : 8 MB
    const size_t oK    = oQ    + (size_t)S_LEN * NH * HD * 2; // Kb : 2 MB
    const size_t oV    = oK    + (size_t)NKV * S_LEN * HD * 2;// Vtb: 2 MB
    bf16* Hb    = (bf16*)(ws + oH);
    bf16* Wqkvb = (bf16*)(ws + oWqkv);
    bf16* Wob   = (bf16*)(ws + oWo);
    bf16* Qb    = (bf16*)(ws + oQ);
    bf16* Kb    = (bf16*)(ws + oK);
    bf16* Vtb   = (bf16*)(ws + oV);
    bf16* Ab    = Hb;   // alias: hidden bf16 dead after QKV GEMM

    convert_kernel<<<2048, 256, 0, stream>>>(h, wq, wk, wv, wo, Hb, Wqkvb, Wob);

    // QKV: blocks = (2048/256) x (3072/64) = 8 x 48 = 384
    qkv_rope_kernel<<<384, 256, 0, stream>>>(Hb, Wqkvb, cosT, sinT, Qb, Kb, Vtb);

    // attention: 32 heads x 128 query tiles, one wave each
    attn_kernel<<<NH * (S_LEN / 16), 32, 0, stream>>>(Qb, Kb, Vtb, Ab);

    // out proj: blocks = (2048/256) x (2048/64) = 8 x 32 = 256
    out_gemm_kernel<<<256, 256, 0, stream>>>(Ab, Wob, out);
}
